// Attention_55697135894780
// MI455X (gfx1250) — matrix operand order repaired and checked by reading, not compiled, not run on hardware
//
#include <hip/hip_runtime.h>
#include <hip/hip_bf16.h>

// Problem constants (from reference): x [2, 64, 12, 16, 16]
#define BATCH 2
#define C_CH  64
#define NPOS  3072      // 12*16*16
#define NHEAD 4
#define DH    16
#define EPSV  1e-5f

typedef __attribute__((ext_vector_type(16))) _Float16 v16h;
typedef __attribute__((ext_vector_type(8)))  _Float16 v8h;
typedef __attribute__((ext_vector_type(8)))  float    v8f;

__device__ __forceinline__ v16h cat8(v8h lo, v8h hi) {
  return __builtin_shufflevector(lo, hi, 0, 1, 2, 3, 4, 5, 6, 7,
                                 8, 9, 10, 11, 12, 13, 14, 15);
}

// ---------------------------------------------------------------------------
// Contiguous WMMA fragment loaders (CDNA5 wave32, CDNA5 ISA §7.12.2)
// K must be the unit-stride memory dimension; all data layouts below are
// chosen so this holds -> every fragment is one/two b128 loads per lane.
//
// A 16x32 f16 frag: lane row M = lane&15; halves 0..7 -> K = kb..kb+7,
//                   halves 8..15 -> K = 16+kb..23+kb,  kb = (lane>>4)*8.
// B 32x16 f16 frag: lane col N = lane&15; same K order as A:
//                   halves 0..7 -> K = kb..kb+7, halves 8..15 -> K = 16+kb..23+kb.
// ---------------------------------------------------------------------------
__device__ __forceinline__ v16h load_a32(const _Float16* __restrict__ p, int ld_row) {
  const int lane = threadIdx.x & 31;
  const _Float16* q = p + (long)(lane & 15) * ld_row + ((lane >> 4) << 3);
  v8h lo = *(const v8h*)q;
  v8h hi = *(const v8h*)(q + 16);
  return cat8(lo, hi);
}

__device__ __forceinline__ v16h load_a16(const _Float16* __restrict__ p, int ld_row) {
  // K = 16..31 is zero padding (head-dim 16 padded to the K=32 WMMA shape)
  const int lane = threadIdx.x & 31;
  const _Float16* q = p + (long)(lane & 15) * ld_row + ((lane >> 4) << 3);
  v8h lo = *(const v8h*)q;
  v8h hi = {};
  return cat8(lo, hi);
}

__device__ __forceinline__ v16h load_b32(const _Float16* __restrict__ p, long ld_col) {
  const int lane = threadIdx.x & 31;
  const _Float16* q = p + (long)(lane & 15) * ld_col + ((lane >> 4) << 3);
  v8h lo = *(const v8h*)q;
  v8h hi = *(const v8h*)(q + 16);
  return cat8(lo, hi);
}

__device__ __forceinline__ v16h load_b16(const _Float16* __restrict__ p, long ld_col) {
  // K = 16..31 zero: halves 8..15 of every lane carry zero-padded K values
  const int lane = threadIdx.x & 31;
  const _Float16* q = p + (long)(lane & 15) * ld_col + ((lane >> 4) << 3);
  v8h lo = *(const v8h*)q;
  v8h hi = {};
  return cat8(lo, hi);
}

__device__ __forceinline__ v8f wmma_f16(v16h a, v16h b, v8f c) {
  return __builtin_amdgcn_wmma_f32_16x16x32_f16(false, a, false, b,
                                                (short)0, c, false, false);
}

// ---------------------------------------------------------------------------
// Kernel 0: convert weights to f16
// ---------------------------------------------------------------------------
__global__ void convert_w_kernel(const float* __restrict__ wqkv,
                                 const float* __restrict__ wout,
                                 _Float16* __restrict__ wq16,
                                 _Float16* __restrict__ wo16) {
  int i = blockIdx.x * blockDim.x + threadIdx.x;
  if (i < 3 * C_CH * C_CH) wq16[i] = (_Float16)wqkv[i];
  if (i < C_CH * C_CH)     wo16[i] = (_Float16)wout[i];
}

// ---------------------------------------------------------------------------
// Kernel 1: GroupNorm -> nrmT f16 [B][NPOS][64]  (channel contiguous)
// ---------------------------------------------------------------------------
__global__ void gn_kernel(const float* __restrict__ x,
                          const float* __restrict__ gw,
                          const float* __restrict__ gb,
                          _Float16* __restrict__ nrmT) {
  const int b = blockIdx.x >> 2;
  const int g = blockIdx.x & 3;
  const long base = ((long)b * C_CH + g * 16) * NPOS;
  const int tid = threadIdx.x;

  float s = 0.f, ss = 0.f;
  for (int e = tid; e < 16 * NPOS; e += 256) {
    float v = x[base + e];
    s += v; ss += v * v;
  }
  __shared__ float sh_s[256], sh_q[256];
  sh_s[tid] = s; sh_q[tid] = ss;
  __syncthreads();
  for (int off = 128; off > 0; off >>= 1) {
    if (tid < off) { sh_s[tid] += sh_s[tid + off]; sh_q[tid] += sh_q[tid + off]; }
    __syncthreads();
  }
  const float inv_cnt = 1.0f / (16.0f * NPOS);
  const float mean = sh_s[0] * inv_cnt;
  const float var  = sh_q[0] * inv_cnt - mean * mean;
  const float rstd = rsqrtf(var + EPSV);

  for (int cl = 0; cl < 16; ++cl) {
    const int c = g * 16 + cl;
    const float w = gw[c] * rstd;
    const float o = gb[c] - mean * w;
    for (int n = tid; n < NPOS; n += 256)
      nrmT[((long)b * NPOS + n) * C_CH + c] = (_Float16)(x[base + cl * NPOS + n] * w + o);
  }
}

// ---------------------------------------------------------------------------
// Kernel 2: QKV projection (M=192, K=64, N=3072 per batch; 1 wave = 16x16 tile)
//   qT, kT : f16 [b*4+h][pos][16]  (head-dim contiguous; D-tile store is one
//                                   packed b128 per lane)
//   vb     : f16 [b*4+h][d][NPOS]  (j contiguous -> P*V B-frag is b128 loads)
// ---------------------------------------------------------------------------
__global__ void qkv_kernel(const _Float16* __restrict__ wq,
                           const _Float16* __restrict__ nrmT,
                           _Float16* __restrict__ qT,
                           _Float16* __restrict__ kT,
                           _Float16* __restrict__ vb) {
  const int wave = (int)((blockIdx.x * (long)blockDim.x + threadIdx.x) >> 5);
  const int ntn = NPOS / 16;                 // 192
  const int b   = wave / (12 * ntn);
  const int r   = wave % (12 * ntn);
  const int mt  = r / ntn;                   // 0..11: q heads 0-3, k 4-7, v 8-11
  const int nt  = r % ntn;
  const int lane = threadIdx.x & 31;

  v8f acc = {};
#pragma unroll
  for (int k0 = 0; k0 < C_CH; k0 += 32) {
    v16h a  = load_a32(wq + (long)(mt * 16) * C_CH + k0, C_CH);
    v16h bm = load_b32(nrmT + ((long)b * NPOS + nt * 16) * C_CH + k0, C_CH);
    acc = wmma_f16(a, bm, acc);
  }
  const int nc   = lane & 15;
  const int half = lane >> 4;
  if (mt < 8) {
    _Float16* dst = (mt < 4) ? qT : kT;
    const int h = mt & 3;
    v8h pk;
#pragma unroll
    for (int rr = 0; rr < 8; ++rr) pk[rr] = (_Float16)acc[rr];   // d = rr + half*8
    *(v8h*)(dst + (((long)(b * 4 + h) * NPOS) + nt * 16 + nc) * DH + half * 8) = pk;
  } else {
    const int h = mt - 8;
#pragma unroll
    for (int rr = 0; rr < 8; ++rr) {
      int d = rr + half * 8;
      vb[((long)(b * 4 + h) * DH + d) * NPOS + nt * 16 + nc] = (_Float16)acc[rr];
    }
  }
}

// ---------------------------------------------------------------------------
// Kernel 3: fused flash-style attention, 1 wave = (b, head, 16 query rows).
// 64 key columns per iteration: 4 sim WMMAs + 2 P*V WMMAs, so each per-row
// shuffle-reduction tree amortizes over 64 columns. All fragments are
// contiguous b128 loads; P staged via a wave-private 16x64 LDS tile (LDS ops
// are in-order per wave -> compile-time wave_barrier only, no s_barrier).
// Output oatT f16 [b][pos][64] (channel contiguous for the projection).
// ---------------------------------------------------------------------------
__global__ void attn_kernel(const _Float16* __restrict__ qT,
                            const _Float16* __restrict__ kT,
                            const _Float16* __restrict__ vb,
                            _Float16* __restrict__ oatT) {
  __shared__ __align__(16) _Float16 Pbuf[4][16][64];
  const int wId  = threadIdx.x >> 5;
  const int lane = threadIdx.x & 31;
  const int task = blockIdx.x * 4 + wId;     // (b*4 + h) * 192 + itile
  const int it = task % (NPOS / 16);
  const int bh = task / (NPOS / 16);
  const int i0 = it * 16;

  const _Float16* qTb = qT + ((long)bh * NPOS + i0) * DH;
  const _Float16* kTb = kT + (long)bh * NPOS * DH;
  const _Float16* vbb = vb + (long)bh * DH * NPOS;

  v16h aQ = load_a16(qTb, DH);               // A(m=i, k=d), d padded to 32

  v8f acc = {};
  float mrow[8], lrow[8];
#pragma unroll
  for (int rr = 0; rr < 8; ++rr) { mrow[rr] = -1e30f; lrow[rr] = 0.f; }

  for (int j0 = 0; j0 < NPOS; j0 += 64) {
    if (j0 + 64 < NPOS) {                    // prefetch next 64-column tile
      __builtin_prefetch(kTb + (long)(j0 + 64) * DH + lane * 32, 0, 1);
      __builtin_prefetch(vbb + (long)(lane & 15) * NPOS + j0 + 64 + (lane >> 4) * 32,
                         0, 1);
    }
    // 4 sim tiles: s[t] covers columns j0 + 16t .. j0 + 16t + 15
    v8f s0, s1, s2, s3;
    {
      v8f z = {};
      s0 = wmma_f16(aQ, load_b16(kTb + (long)(j0     ) * DH, DH), z);
      s1 = wmma_f16(aQ, load_b16(kTb + (long)(j0 + 16) * DH, DH), z);
      s2 = wmma_f16(aQ, load_b16(kTb + (long)(j0 + 32) * DH, DH), z);
      s3 = wmma_f16(aQ, load_b16(kTb + (long)(j0 + 48) * DH, DH), z);
    }

    // Online softmax over 64 columns; row m = rr + (lane>>4)*8 spans the
    // 16 lanes of a half, xor masks 1..8 reduce within each half.
#pragma unroll
    for (int rr = 0; rr < 8; ++rr) {
      float tmax = fmaxf(fmaxf(s0[rr], s1[rr]), fmaxf(s2[rr], s3[rr]));
      for (int msk = 1; msk < 16; msk <<= 1)
        tmax = fmaxf(tmax, __shfl_xor(tmax, msk, 32));
      float mnew  = fmaxf(mrow[rr], tmax);
      float scale = __expf(mrow[rr] - mnew);
      float p0 = __expf(s0[rr] - mnew);
      float p1 = __expf(s1[rr] - mnew);
      float p2 = __expf(s2[rr] - mnew);
      float p3 = __expf(s3[rr] - mnew);
      float psum = (p0 + p1) + (p2 + p3);
      for (int msk = 1; msk < 16; msk <<= 1)
        psum += __shfl_xor(psum, msk, 32);
      lrow[rr] = lrow[rr] * scale + psum;
      mrow[rr] = mnew;
      acc[rr] *= scale;
      int mr = rr + (lane >> 4) * 8;
      int nc = lane & 15;
      Pbuf[wId][mr][nc]      = (_Float16)p0;
      Pbuf[wId][mr][nc + 16] = (_Float16)p1;
      Pbuf[wId][mr][nc + 32] = (_Float16)p2;
      Pbuf[wId][mr][nc + 48] = (_Float16)p3;
    }
    __builtin_amdgcn_wave_barrier();         // keep DS store->load ordered
    // P (16x64) * V (64x16) as two K=32 WMMAs
    v16h aP0 = load_a32(&Pbuf[wId][0][0], 64);
    v16h aP1 = load_a32(&Pbuf[wId][0][32], 64);
    v16h bV0 = load_b32(vbb + j0, NPOS);
    v16h bV1 = load_b32(vbb + j0 + 32, NPOS);
    acc = wmma_f16(aP0, bV0, acc);
    acc = wmma_f16(aP1, bV1, acc);
    __builtin_amdgcn_wave_barrier();
  }

  const int b  = bh >> 2;
  const int h  = bh & 3;
  const int nc = lane & 15;                  // d
  const int half = lane >> 4;
#pragma unroll
  for (int rr = 0; rr < 8; ++rr) {
    int m = rr + half * 8;                   // i - i0
    oatT[((long)b * NPOS + i0 + m) * C_CH + h * 16 + nc] =
        (_Float16)(acc[rr] / lrow[rr]);
  }
}

// ---------------------------------------------------------------------------
// Kernel 4: output projection + bias + residual (f32 out, reference layout)
// ---------------------------------------------------------------------------
__global__ void proj_kernel(const _Float16* __restrict__ wo,
                            const _Float16* __restrict__ oatT,
                            const float* __restrict__ bout,
                            const float* __restrict__ x,
                            float* __restrict__ out) {
  const int wave = (int)((blockIdx.x * (long)blockDim.x + threadIdx.x) >> 5);
  const int ntn = NPOS / 16;
  const int b   = wave / (4 * ntn);
  const int r   = wave % (4 * ntn);
  const int mt  = r / ntn;
  const int nt  = r % ntn;
  const int lane = threadIdx.x & 31;

  v8f acc = {};
#pragma unroll
  for (int k0 = 0; k0 < C_CH; k0 += 32) {
    v16h a  = load_a32(wo + (long)(mt * 16) * C_CH + k0, C_CH);
    v16h bm = load_b32(oatT + ((long)b * NPOS + nt * 16) * C_CH + k0, C_CH);
    acc = wmma_f16(a, bm, acc);
  }
  const int nc   = lane & 15;
  const int half = lane >> 4;
#pragma unroll
  for (int rr = 0; rr < 8; ++rr) {
    int m = rr + half * 8;
    int o = mt * 16 + m;
    long idx = ((long)b * C_CH + o) * NPOS + nt * 16 + nc;
    out[idx] = acc[rr] + bout[o] + x[idx];
  }
}

// ---------------------------------------------------------------------------
extern "C" void kernel_launch(void* const* d_in, const int* in_sizes, int n_in,
                              void* d_out, int out_size, void* d_ws, size_t ws_size,
                              hipStream_t stream) {
  const float* x    = (const float*)d_in[0];   // [2,64,12,16,16]
  const float* gnw  = (const float*)d_in[1];   // [64]
  const float* gnb  = (const float*)d_in[2];   // [64]
  const float* wqkv = (const float*)d_in[3];   // [192,64]
  const float* wout = (const float*)d_in[4];   // [64,64]
  const float* bout = (const float*)d_in[5];   // [64]
  float* out = (float*)d_out;

  // Workspace (f16): nrmT | qT | kT | vb | oatT | wq16 | wo16  (~4 MB)
  _Float16* nrmT = (_Float16*)d_ws;                       // B*NPOS*64
  _Float16* qT   = nrmT + (long)BATCH * NPOS * C_CH;      // B*4*NPOS*16
  _Float16* kT   = qT   + (long)BATCH * NHEAD * NPOS * DH;
  _Float16* vb   = kT   + (long)BATCH * NHEAD * NPOS * DH;
  _Float16* oatT = vb   + (long)BATCH * NHEAD * DH * NPOS;
  _Float16* wq16 = oatT + (long)BATCH * NPOS * C_CH;
  _Float16* wo16 = wq16 + 3 * C_CH * C_CH;

  convert_w_kernel<<<(3 * C_CH * C_CH + 255) / 256, 256, 0, stream>>>(
      wqkv, wout, wq16, wo16);

  gn_kernel<<<BATCH * 4, 256, 0, stream>>>(x, gnw, gnb, nrmT);

  qkv_kernel<<<(BATCH * 12 * (NPOS / 16)) / 4, 128, 0, stream>>>(
      wq16, nrmT, qT, kT, vb);

  attn_kernel<<<(BATCH * NHEAD * (NPOS / 16)) / 4, 128, 0, stream>>>(
      qT, kT, vb, oatT);

  proj_kernel<<<(BATCH * 4 * (NPOS / 16)) / 4, 128, 0, stream>>>(
      wo16, oatT, bout, x, out);
}
